// GIN_14302241096445
// MI455X (gfx1250) — matrix operand order repaired and checked by reading, not compiled, not run on hardware
//
#include <hip/hip_runtime.h>
#include <hip/hip_bf16.h>

// ---------------------------------------------------------------------------
// GIN forward on MI455X (gfx1250, wave32).
//  - GEMMs run on v_wmma_f32_16x16x32_bf16 (f32 accumulate), A fragments
//    preloaded for the whole K dim, B fragments software-pipelined.
//  - Edge aggregation / graph pooling use f32 global atomics (L2-resident:
//    feature maps are 51.2 MB vs 192 MB L2).
//  - Weights are pre-swizzled once into the per-lane WMMA B-operand layout.
// ---------------------------------------------------------------------------

typedef __attribute__((ext_vector_type(16))) __bf16 v16bf;
typedef __attribute__((ext_vector_type(8)))  float  v8f;
typedef __attribute__((ext_vector_type(4)))  float  v4f;

#define BN_EPS 1e-5f

// ---------------------------------------------------------------------------
// Pre-swizzle a fp32 [K,N] row-major weight matrix into bf16 WMMA B fragments.
// Fragment f = ntile*KC + kc holds the 32x16 tile for output cols
// [ntile*16, ntile*16+16) and K rows [kc*32, kc*32+32).
// Per-lane layout (16x16x32 bf16 B operand, wave32):
//   lane 0..15 : col n = lane,    elements j=0..15 -> k = kc*32 + j
//   lane 16..31: col n = lane-16, elements j=0..15 -> k = kc*32 + 16 + j
// Stored so each lane's 16 bf16 values are 32 contiguous bytes.
// ---------------------------------------------------------------------------
__global__ void prep_frags(const float* __restrict__ W, __bf16* __restrict__ F,
                           int K, int N) {
    const int KC = K >> 5;
    const size_t total = (size_t)K * (size_t)N;
    for (size_t t = (size_t)blockIdx.x * blockDim.x + threadIdx.x; t < total;
         t += (size_t)gridDim.x * blockDim.x) {
        int f    = (int)(t >> 9);        // / (32 lanes * 16 elems)
        int rem  = (int)(t & 511);
        int lane = rem >> 4;
        int j    = rem & 15;
        int ntile = f / KC;
        int kc    = f - ntile * KC;
        int n = ntile * 16 + (lane & 15);
        int k = kc * 32 + ((lane >> 4) ? 16 : 0) + j;
        F[t] = (__bf16)W[(size_t)k * N + n];
    }
}

// ---------------------------------------------------------------------------
// Vectorized copy: dst = src  (one float4 per thread)
// ---------------------------------------------------------------------------
__global__ void copy_f4(const float* __restrict__ src, float* __restrict__ dst,
                        long n4) {
    long t = (long)blockIdx.x * blockDim.x + threadIdx.x;
    if (t < n4) {
        ((float4*)dst)[t] = ((const float4*)src)[t];
    }
}

// ---------------------------------------------------------------------------
// Edge scatter-add: agg[dst[e]][:] += x[src[e]][:]
// One wave per edge, float4 per lane (32*4 = 128 features). The random
// gathers and f32 atomics resolve in L2.
// ---------------------------------------------------------------------------
__global__ void edge_agg(const float* __restrict__ x, const int* __restrict__ src,
                         const int* __restrict__ dst, float* __restrict__ agg,
                         int E) {
    int t    = blockIdx.x * blockDim.x + threadIdx.x;
    int e    = t >> 5;
    int lane = t & 31;
    if (e >= E) return;
    int s = src[e];
    int d = dst[e];
    float4 v = *(const float4*)(x + (size_t)s * 128 + lane * 4);
    float* ap = agg + (size_t)d * 128 + lane * 4;
    atomicAdd(ap + 0, v.x);
    atomicAdd(ap + 1, v.y);
    atomicAdd(ap + 2, v.z);
    atomicAdd(ap + 3, v.w);
}

// ---------------------------------------------------------------------------
// WMMA GEMM (K = KC*32 compile-time): C[:, colBase:colBase+128] =
//     act( (A @ B + bias) * (gamma*rsqrt(1+eps)) + beta )
// A: fp32 [M, KC*32] row-major, converted to bf16 on load; all KC A-fragments
// preloaded into registers. B fragments are double-buffered so each WMMA
// overlaps the next tile's global_load_b128 pair.
// Block = 256 threads = 8 waves; wave w computes rows
// [blockIdx.x*128 + w*16, +16) x 128 output cols. colBase = blockIdx.y*128.
// ---------------------------------------------------------------------------
template <int KC>
__global__ __launch_bounds__(256)
void wmma_gemm(const float* __restrict__ A, const __bf16* __restrict__ Bf,
               float* __restrict__ C, int M, int Nout,
               const float* __restrict__ bias, const float* __restrict__ gamma,
               const float* __restrict__ beta, int relu) {
    const int wave    = threadIdx.x >> 5;
    const int lane    = threadIdx.x & 31;
    const int halfSel = lane >> 4;   // 0: lanes 0-15, 1: lanes 16-31
    const int lrow    = lane & 15;
    const int rowBase = blockIdx.x * 128 + wave * 16;
    if (rowBase >= M) return;        // wave-uniform exit; EXEC stays full
    const int colBase = blockIdx.y * 128;

    int arow = rowBase + lrow;
    if (arow >= M) arow = M - 1;     // clamp loads; stores are guarded below
    const float* aptr = A + (size_t)arow * (KC * 32);

    // --- preload ALL A fragments (16x32 bf16 each), contiguous k order (same as B):
    //  lanes 0-15 : row=lane,    elems 0..15 -> k=kc*32+0..15
    //  lanes 16-31: row=lane-16, elems 0..15 -> k=kc*32+16..31
    v16bf afr[KC];
#pragma unroll
    for (int kc = 0; kc < KC; ++kc) {
        const int c0 = kc * 32 + (halfSel ? 16 : 0);
        v4f a0 = *(const v4f*)(aptr + c0);
        v4f a1 = *(const v4f*)(aptr + c0 + 4);
        v4f a2 = *(const v4f*)(aptr + c0 + 8);
        v4f a3 = *(const v4f*)(aptr + c0 + 12);
#pragma unroll
        for (int j = 0; j < 4; ++j) {
            afr[kc][j]      = (__bf16)a0[j];
            afr[kc][4 + j]  = (__bf16)a1[j];
            afr[kc][8 + j]  = (__bf16)a2[j];
            afr[kc][12 + j] = (__bf16)a3[j];
        }
    }

    auto bload = [&](int nt, int kc) -> v16bf {
        const int ntg = (colBase >> 4) + nt;
        return *(const v16bf*)(Bf + (((size_t)ntg * KC + kc) * 32 + lane) * 16);
    };

    v8f acc[8];
#pragma unroll
    for (int i = 0; i < 8; ++i)
#pragma unroll
        for (int j = 0; j < 8; ++j) acc[i][j] = 0.0f;

    // --- software-pipelined main loop: issue load for tile (nt+1) before
    //     the WMMA of tile nt, so the XDL op hides the B fetch latency.
    v16bf bcur = bload(0, 0);
#pragma unroll
    for (int kc = 0; kc < KC; ++kc) {
#pragma unroll
        for (int nt = 0; nt < 8; ++nt) {
            int nn = nt + 1, kk = kc;
            if (nn == 8) { nn = 0; ++kk; }
            v16bf bnxt = bcur;
            if (kk < KC) bnxt = bload(nn, kk);
            acc[nt] = __builtin_amdgcn_wmma_f32_16x16x32_bf16(
                false, afr[kc], false, bcur, (short)0, acc[nt], false, false);
            bcur = bnxt;
        }
    }

    // --- epilogue: (acc + bias) * gamma*rsqrt(1+eps) + beta, optional ReLU.
    // D layout: lanes 0-15: col=lane, vgpr v -> row v; lanes 16-31: row v+8.
#pragma unroll
    for (int nt = 0; nt < 8; ++nt) {
        const int col = colBase + nt * 16 + lrow;
        const float s  = gamma ? gamma[col] * rsqrtf(1.0f + BN_EPS) : 1.0f;
        const float bb = bias ? bias[col] : 0.0f;
        const float be = beta ? beta[col] : 0.0f;
#pragma unroll
        for (int v = 0; v < 8; ++v) {
            const int r = rowBase + (halfSel ? v + 8 : v);
            if (r < M) {
                float val = (acc[nt][v] + bb) * s + be;
                if (relu) val = fmaxf(val, 0.0f);
                C[(size_t)r * Nout + col] = val;
            }
        }
    }
}

// ---------------------------------------------------------------------------
// Graph pooling of all three layers into a pre-concatenated [G, 384] buffer.
// One wave per node; float4 per lane.
// ---------------------------------------------------------------------------
__global__ void pool3(const float* __restrict__ h1, const float* __restrict__ h2,
                      const float* __restrict__ h3, const int* __restrict__ batch,
                      float* __restrict__ P, int Nn) {
    int t    = blockIdx.x * blockDim.x + threadIdx.x;
    int node = t >> 5;
    int lane = t & 31;
    if (node >= Nn) return;
    int g = batch[node];
    float* p = P + (size_t)g * 384 + lane * 4;
    float4 v1 = *(const float4*)(h1 + (size_t)node * 128 + lane * 4);
    float4 v2 = *(const float4*)(h2 + (size_t)node * 128 + lane * 4);
    float4 v3 = *(const float4*)(h3 + (size_t)node * 128 + lane * 4);
    atomicAdd(p + 0, v1.x);   atomicAdd(p + 1, v1.y);
    atomicAdd(p + 2, v1.z);   atomicAdd(p + 3, v1.w);
    atomicAdd(p + 128 + 0, v2.x); atomicAdd(p + 128 + 1, v2.y);
    atomicAdd(p + 128 + 2, v2.z); atomicAdd(p + 128 + 3, v2.w);
    atomicAdd(p + 256 + 0, v3.x); atomicAdd(p + 256 + 1, v3.y);
    atomicAdd(p + 256 + 2, v3.z); atomicAdd(p + 256 + 3, v3.w);
}

// ---------------------------------------------------------------------------
// Final tiny GEMM: out[512,10] = Q[512,384] @ W[384,10] + b
// ---------------------------------------------------------------------------
__global__ void lin2_kernel(const float* __restrict__ Q, const float* __restrict__ W,
                            const float* __restrict__ b, float* __restrict__ out,
                            int G) {
    int t = blockIdx.x * blockDim.x + threadIdx.x;
    if (t >= G * 10) return;
    int r = t / 10, c = t % 10;
    float s = b[c];
    const float* q = Q + (size_t)r * 384;
    for (int k = 0; k < 384; ++k) s += q[k] * W[k * 10 + c];
    out[t] = s;
}

// ---------------------------------------------------------------------------
extern "C" void kernel_launch(void* const* d_in, const int* in_sizes, int n_in,
                              void* d_out, int out_size, void* d_ws, size_t ws_size,
                              hipStream_t stream) {
    const float* x     = (const float*)d_in[0];
    const int*   ei    = (const int*)d_in[1];
    const int*   batch = (const int*)d_in[2];
    const float* w1[3]    = {(const float*)d_in[3],  (const float*)d_in[9],  (const float*)d_in[15]};
    const float* b1[3]    = {(const float*)d_in[4],  (const float*)d_in[10], (const float*)d_in[16]};
    const float* gamma[3] = {(const float*)d_in[5],  (const float*)d_in[11], (const float*)d_in[17]};
    const float* beta[3]  = {(const float*)d_in[6],  (const float*)d_in[12], (const float*)d_in[18]};
    const float* w2[3]    = {(const float*)d_in[7],  (const float*)d_in[13], (const float*)d_in[19]};
    const float* b2[3]    = {(const float*)d_in[8],  (const float*)d_in[14], (const float*)d_in[20]};
    const float* lin1_w = (const float*)d_in[21];
    const float* lin1_b = (const float*)d_in[22];
    const float* lin2_w = (const float*)d_in[23];
    const float* lin2_b = (const float*)d_in[24];

    const int Nn = in_sizes[0] / 128;   // 100000
    const int E  = in_sizes[1] / 2;     // 1600000
    const int G  = 512;

    // --- workspace layout ---
    char* ws = (char*)d_ws;
    size_t off = 0;
    auto alloc = [&](size_t bytes) -> void* {
        void* p = ws + off;
        off += (bytes + 255) & ~(size_t)255;
        return p;
    };
    float* agg = (float*)alloc((size_t)Nn * 128 * 4);
    float* tmp = (float*)alloc((size_t)Nn * 128 * 4);
    float* h[3];
    for (int i = 0; i < 3; ++i) h[i] = (float*)alloc((size_t)Nn * 128 * 4);
    __bf16* fw1[3];
    __bf16* fw2[3];
    for (int i = 0; i < 3; ++i) {
        fw1[i] = (__bf16*)alloc(128 * 128 * 2);
        fw2[i] = (__bf16*)alloc(128 * 128 * 2);
    }
    __bf16* flin1 = (__bf16*)alloc(384 * 384 * 2);
    float* P = (float*)alloc((size_t)G * 384 * 4);
    float* Q = (float*)alloc((size_t)G * 384 * 4);

    // --- weight pre-swizzle (tiny) ---
    for (int i = 0; i < 3; ++i) {
        prep_frags<<<64, 256, 0, stream>>>(w1[i], fw1[i], 128, 128);
        prep_frags<<<64, 256, 0, stream>>>(w2[i], fw2[i], 128, 128);
    }
    prep_frags<<<128, 256, 0, stream>>>(lin1_w, flin1, 384, 384);

    hipMemsetAsync(P, 0, (size_t)G * 384 * 4, stream);

    const int* src = ei;
    const int* dst = ei + E;
    const long n4 = (long)Nn * 32;                 // float4 count per feature map
    const int rowBlocks = (Nn + 127) / 128;

    const float* in = x;
    for (int l = 0; l < 3; ++l) {
        // agg = in ; agg += scatter(in[src] -> dst)
        copy_f4<<<(int)((n4 + 255) / 256), 256, 0, stream>>>(in, agg, n4);
        edge_agg<<<(E * 32 + 255) / 256, 256, 0, stream>>>(in, src, dst, agg, E);
        // tmp = relu( (agg @ W1 + b1) * gamma/sqrt(1+eps) + beta )
        wmma_gemm<4><<<dim3(rowBlocks, 1), 256, 0, stream>>>(
            agg, fw1[l], tmp, Nn, 128, b1[l], gamma[l], beta[l], 1);
        // h[l] = relu( tmp @ W2 + b2 )
        wmma_gemm<4><<<dim3(rowBlocks, 1), 256, 0, stream>>>(
            tmp, fw2[l], h[l], Nn, 128, b2[l], nullptr, nullptr, 1);
        in = h[l];
    }

    pool3<<<(Nn * 32 + 255) / 256, 256, 0, stream>>>(h[0], h[1], h[2], batch, P, Nn);

    // Q = relu( P @ lin1_w + lin1_b )   (512 x 384, K=384)
    wmma_gemm<12><<<dim3(4, 3), 256, 0, stream>>>(
        P, flin1, Q, G, 384, lin1_b, nullptr, nullptr, 1);

    lin2_kernel<<<(G * 10 + 255) / 256, 256, 0, stream>>>(
        Q, lin2_w, lin2_b, (float*)d_out, G);
}
